// GQAttention_1941325218431
// MI455X (gfx1250) — matrix operand order repaired and checked by reading, not compiled, not run on hardware
//
#include <hip/hip_runtime.h>
#include <hip/hip_bf16.h>
#include <math.h>

typedef __attribute__((ext_vector_type(16))) _Float16 v16h;
typedef __attribute__((ext_vector_type(8)))  _Float16 v8h;
typedef __attribute__((ext_vector_type(4)))  _Float16 v4h;
typedef __attribute__((ext_vector_type(8)))  float    v8f;

#define B_    2
#define T_    2048
#define D_    2048
#define H_    32
#define KV_   8
#define HD_   64
#define NREP_ 4

#ifndef USE_ASYNC_LDS
#define USE_ASYNC_LDS 1
#endif

// ---------------------------------------------------------------------------
// CDNA5 async global->LDS DMA (ASYNCcnt-tracked). VDST = per-lane LDS byte
// offset (low 32 bits of the flat LDS address), VADDR = 64-bit global addr.
// ---------------------------------------------------------------------------
__device__ __forceinline__ void async_copy_b128(void* lds_dst, const void* gsrc) {
#if USE_ASYNC_LDS
  const unsigned ldsoff = (unsigned)(uintptr_t)lds_dst;
  asm volatile("global_load_async_to_lds_b128 %0, %1, off"
               :: "v"(ldsoff), "v"(gsrc) : "memory");
#else
  *(v8h*)lds_dst = *(const v8h*)gsrc;
#endif
}
__device__ __forceinline__ void async_wait0() {
#if USE_ASYNC_LDS
  asm volatile("s_wait_asynccnt 0x0" ::: "memory");
#endif
}

// ---------------------------------------------------------------------------
// CDNA5 wave32 WMMA fragment layouts (16-bit):
//  A 16x32 (MxK): lane m=lane%16; lanes<16: K=0..7 / 16..23; lanes>=16:
//                 K=8..15 / 24..31  -> two contiguous 16B chunks per lane.
//  B 32x16 (KxN): lane n=lane%16; lanes<16: K=0..7 / 16..23; lanes>=16:
//                 K=8..15 / 24..31  -> two contiguous 16B chunks when staged [n][k].
//  C/D 16x16 f32: VGPR r -> M = r + 8*(lane/16); N = lane%16.
// ---------------------------------------------------------------------------
__device__ __forceinline__ v16h frag16(const _Float16* plo, const _Float16* phi) {
  const v8h lo = *(const v8h*)plo;
  const v8h hi = *(const v8h*)phi;
  return __builtin_shufflevector(lo, hi, 0, 1, 2, 3, 4, 5, 6, 7,
                                 8, 9, 10, 11, 12, 13, 14, 15);
}

__device__ __forceinline__ v16h load_A_frag(const _Float16* base, int m0, int k0, int ld) {
  const int lane = threadIdx.x & 31;
  const _Float16* p = base + (m0 + (lane & 15)) * ld + k0 + (lane >> 4) * 8;
  return frag16(p, p + 16);
}

__device__ __forceinline__ v16h load_B_frag(const _Float16* base, int n0, int k0, int ld) {
  const int lane = threadIdx.x & 31;
  const _Float16* p = base + (n0 + (lane & 15)) * ld + k0 + (lane >> 4) * 8;
  return frag16(p, p + 16);
}

__device__ __forceinline__ v8f wmma_f16(v16h a, v16h b, v8f c) {
  return __builtin_amdgcn_wmma_f32_16x16x32_f16(false, a, false, b, (short)0, c, false, false);
}

// ---------------------------------------------------------------------------
// GEMM: C[M,N] = A[M,K] @ B[K,N].  A is fp32 (convert-stage) or f16
// (async DMA stage).  B is fp32 weights (convert + transpose to [n][k]).
// C is fp32 or f16.  Block tile 128x128, K-step 32, 8 waves, wave tile 64x32.
// ---------------------------------------------------------------------------
#define GTM 128
#define GTN 128
#define GTK 32
#define LDA_S 40
#define LDB_S 40

template <typename AT, typename CT>
__global__ __launch_bounds__(256) void gemm_wmma_kern(const AT* __restrict__ A,
                                                      const float* __restrict__ Bm,
                                                      CT* __restrict__ C,
                                                      int M, int N, int K) {
  __shared__ __align__(16) _Float16 As[GTM * LDA_S];  // [m][k]
  __shared__ __align__(16) _Float16 Bs[GTN * LDB_S];  // [n][k]
  const int tid  = threadIdx.x;
  const int lane = tid & 31;
  const int wave = tid >> 5;
  const int wm = wave >> 2;
  const int wn = wave & 3;
  const int m0blk = blockIdx.y * GTM;
  const int n0blk = blockIdx.x * GTN;

  v8f acc[4][2];
#pragma unroll
  for (int i = 0; i < 4; ++i)
#pragma unroll
    for (int j = 0; j < 2; ++j)
#pragma unroll
      for (int r = 0; r < 8; ++r) acc[i][j][r] = 0.0f;

  const int bn4 = tid >> 3;  // 0..31 (n block of 4)
  const int bk4 = tid & 7;   // 0..7  (k block of 4)

  for (int kb = 0; kb < K; kb += GTK) {
    // ---- Stage A tile (128x32) ----
    if constexpr (sizeof(AT) == 2) {
      // f16 source: straight async DMA, 512 16B chunks, 2 per thread
#pragma unroll
      for (int v = 0; v < 2; ++v) {
        const int e8 = tid + v * 256;
        const int r = e8 >> 2;
        const int c = (e8 & 3) * 8;
        async_copy_b128(&As[r * LDA_S + c],
                        (const _Float16*)A + (size_t)(m0blk + r) * K + kb + c);
      }
    } else {
      // fp32 source: load float4, convert, store v4h
#pragma unroll
      for (int v = 0; v < 4; ++v) {
        const int e = (tid + v * 256) * 4;
        const int r = e >> 5;
        const int c = e & 31;
        const float4 f =
            *reinterpret_cast<const float4*>((const float*)A + (size_t)(m0blk + r) * K + kb + c);
        *(v4h*)&As[r * LDA_S + c] =
            (v4h){(_Float16)f.x, (_Float16)f.y, (_Float16)f.z, (_Float16)f.w};
      }
    }
    // ---- Stage B tile (32x128) fp32 -> f16, transposed to [n][k] ----
    {
      const int gk = kb + bk4 * 4;
      const size_t gcol = (size_t)n0blk + bn4 * 4;
      const float4 r0 = *reinterpret_cast<const float4*>(&Bm[(size_t)(gk + 0) * N + gcol]);
      const float4 r1 = *reinterpret_cast<const float4*>(&Bm[(size_t)(gk + 1) * N + gcol]);
      const float4 r2 = *reinterpret_cast<const float4*>(&Bm[(size_t)(gk + 2) * N + gcol]);
      const float4 r3 = *reinterpret_cast<const float4*>(&Bm[(size_t)(gk + 3) * N + gcol]);
      _Float16* dst = &Bs[(bn4 * 4) * LDB_S + bk4 * 4];
      *(v4h*)(dst + 0 * LDB_S) = (v4h){(_Float16)r0.x, (_Float16)r1.x, (_Float16)r2.x, (_Float16)r3.x};
      *(v4h*)(dst + 1 * LDB_S) = (v4h){(_Float16)r0.y, (_Float16)r1.y, (_Float16)r2.y, (_Float16)r3.y};
      *(v4h*)(dst + 2 * LDB_S) = (v4h){(_Float16)r0.z, (_Float16)r1.z, (_Float16)r2.z, (_Float16)r3.z};
      *(v4h*)(dst + 3 * LDB_S) = (v4h){(_Float16)r0.w, (_Float16)r1.w, (_Float16)r2.w, (_Float16)r3.w};
    }
    if constexpr (sizeof(AT) == 2) async_wait0();
    __syncthreads();

    v16h bfr[2];
#pragma unroll
    for (int j = 0; j < 2; ++j) bfr[j] = load_B_frag(Bs, wn * 32 + j * 16, 0, LDB_S);
#pragma unroll
    for (int i = 0; i < 4; ++i) {
      const v16h afr = load_A_frag(As, wm * 64 + i * 16, 0, LDA_S);
#pragma unroll
      for (int j = 0; j < 2; ++j) acc[i][j] = wmma_f16(afr, bfr[j], acc[i][j]);
    }
    __syncthreads();
  }

  const int half = lane >> 4;
  const int nlo  = lane & 15;
#pragma unroll
  for (int i = 0; i < 4; ++i)
#pragma unroll
    for (int j = 0; j < 2; ++j) {
      const int n = n0blk + wn * 32 + j * 16 + nlo;
      const int mbase = m0blk + wm * 64 + i * 16 + 8 * half;
      CT* p = C + (size_t)mbase * N + n;
#pragma unroll
      for (int r = 0; r < 8; ++r) {
        if constexpr (sizeof(CT) == 2)
          p[(size_t)r * N] = (_Float16)acc[i][j][r];
        else
          p[(size_t)r * N] = acc[i][j][r];
      }
    }
}

// ---------------------------------------------------------------------------
// RoPE (interleaved pairs), in place on f16 (rows, nh, HD).
// ---------------------------------------------------------------------------
__global__ void rope_kernel(_Float16* __restrict__ t, const float* __restrict__ fc,
                            const float* __restrict__ fs, int rows, int nh) {
  const int idx = blockIdx.x * blockDim.x + threadIdx.x;
  const int total = rows * nh * (HD_ / 2);
  if (idx >= total) return;
  const int p = idx & 31;
  const int h = (idx >> 5) % nh;
  const int r = idx / (32 * nh);
  const int tpos = r % T_;
  const float c = fc[tpos * 32 + p];
  const float s = fs[tpos * 32 + p];
  _Float16* base = t + ((size_t)r * nh + h) * HD_ + 2 * p;
  const float x0 = (float)base[0], x1 = (float)base[1];
  base[0] = (_Float16)(x0 * c - x1 * s);
  base[1] = (_Float16)(x0 * s + x1 * c);
}

// ---------------------------------------------------------------------------
// Flash-style causal GQA attention (all activations f16, fp32 accum).
// Grid: (T/64, H, B). Block: 128 threads = 4 waves; each wave owns 16 q-rows.
// Q fragments: direct global b128 loads (A layout == contiguous 16B chunks).
// K tile: async DMA to LDS (natural [key][hd] == fragment-major for QK^T).
// V tile: transposed to [hd][key] (fragment-major for PV).
// ---------------------------------------------------------------------------
#define AT_LD 72

__global__ __launch_bounds__(128) void attn_kernel(const _Float16* __restrict__ Q,
                                                   const _Float16* __restrict__ Kg,
                                                   const _Float16* __restrict__ Vg,
                                                   _Float16* __restrict__ O) {
  __shared__ __align__(16) _Float16 Ks[64 * AT_LD];      // [key][hd]
  __shared__ __align__(16) _Float16 Vt[HD_ * AT_LD];     // [hd][key]
  __shared__ __align__(16) _Float16 Ps[4 * 16 * AT_LD];  // per-wave P [q][key]

  const int tid = threadIdx.x, lane = tid & 31, wave = tid >> 5;
  const int half = lane >> 4, nlo = lane & 15;
  const int qtile = blockIdx.x;
  const int h = blockIdx.y;
  const int b = blockIdx.z;
  const int kvh = h / NREP_;
  const int qbase = qtile * 64;
  const int qrow0 = qbase + wave * 16;

  // Q fragments straight from global (per-lane 16B chunks)
  const _Float16* qrow = &Q[(((size_t)b * T_ + qrow0 + nlo) * H_ + h) * HD_];
  v16h aq[2];
#pragma unroll
  for (int ks = 0; ks < 2; ++ks) {
    const _Float16* p = qrow + ks * 32 + half * 8;
    aq[ks] = frag16(p, p + 16);
  }

  v8f oacc[4];
#pragma unroll
  for (int j = 0; j < 4; ++j)
#pragma unroll
    for (int r = 0; r < 8; ++r) oacc[j][r] = 0.0f;
  float lsum[8], mmax[8];
#pragma unroll
  for (int i = 0; i < 8; ++i) { lsum[i] = 0.0f; mmax[i] = -1e30f; }

  const int nkb = qtile + 1;  // causal
  for (int kb = 0; kb < nkb; ++kb) {
    const int kbase = kb * 64;

    // ---- K tile: async DMA, natural layout, 512 16B chunks / 128 thr ----
#pragma unroll
    for (int v = 0; v < 4; ++v) {
      const int e8 = tid + v * 128;
      const int key = e8 >> 3;
      const int c = (e8 & 7) * 8;
      async_copy_b128(&Ks[key * AT_LD + c],
                      &Kg[(((size_t)b * T_ + kbase + key) * KV_ + kvh) * HD_ + c]);
    }
    // ---- V tile: 4x4 block transpose to [hd][key] ----
#pragma unroll
    for (int bi = 0; bi < 2; ++bi) {
      const int blk = tid + bi * 128;
      const int bk4 = blk & 15;   // key block
      const int bh4 = blk >> 4;   // hd block
      const int gk = kbase + bk4 * 4;
      const int gh = bh4 * 4;
      const v4h r0 = *(const v4h*)&Vg[(((size_t)b * T_ + gk + 0) * KV_ + kvh) * HD_ + gh];
      const v4h r1 = *(const v4h*)&Vg[(((size_t)b * T_ + gk + 1) * KV_ + kvh) * HD_ + gh];
      const v4h r2 = *(const v4h*)&Vg[(((size_t)b * T_ + gk + 2) * KV_ + kvh) * HD_ + gh];
      const v4h r3 = *(const v4h*)&Vg[(((size_t)b * T_ + gk + 3) * KV_ + kvh) * HD_ + gh];
      _Float16* dst = &Vt[(bh4 * 4) * AT_LD + bk4 * 4];
      *(v4h*)(dst + 0 * AT_LD) = (v4h){r0[0], r1[0], r2[0], r3[0]};
      *(v4h*)(dst + 1 * AT_LD) = (v4h){r0[1], r1[1], r2[1], r3[1]};
      *(v4h*)(dst + 2 * AT_LD) = (v4h){r0[2], r1[2], r2[2], r3[2]};
      *(v4h*)(dst + 3 * AT_LD) = (v4h){r0[3], r1[3], r2[3], r3[3]};
    }
    async_wait0();
    __syncthreads();

    // ---- S = Q @ K^T ----
    v8f s[4];
#pragma unroll
    for (int j = 0; j < 4; ++j)
#pragma unroll
      for (int r = 0; r < 8; ++r) s[j][r] = 0.0f;
#pragma unroll
    for (int ks = 0; ks < 2; ++ks)
#pragma unroll
      for (int j = 0; j < 4; ++j) {
        const v16h bk = load_B_frag(Ks, j * 16, ks * 32, AT_LD);
        s[j] = wmma_f16(aq[ks], bk, s[j]);
      }

    // ---- Scale, causal mask, online softmax ----
    float pm[4][8];
#pragma unroll
    for (int i = 0; i < 8; ++i) {
      const int qg = qbase + wave * 16 + i + 8 * half;
      float mx = -1e30f;
#pragma unroll
      for (int j = 0; j < 4; ++j) {
        const int kg = kbase + j * 16 + nlo;
        float vsc = s[j][i] * 0.125f;  // 1/sqrt(64)
        if (kg > qg) vsc = -1e30f;
        pm[j][i] = vsc;
        mx = fmaxf(mx, vsc);
      }
#pragma unroll
      for (int d = 1; d < 16; d <<= 1) mx = fmaxf(mx, __shfl_xor(mx, d, 32));
      const float newm = fmaxf(mmax[i], mx);
      const float corr = __expf(mmax[i] - newm);
      mmax[i] = newm;
      float rsum = 0.0f;
#pragma unroll
      for (int j = 0; j < 4; ++j) {
        pm[j][i] = __expf(pm[j][i] - newm);
        rsum += pm[j][i];
      }
#pragma unroll
      for (int d = 1; d < 16; d <<= 1) rsum += __shfl_xor(rsum, d, 32);
      lsum[i] = lsum[i] * corr + rsum;
#pragma unroll
      for (int j = 0; j < 4; ++j) oacc[j][i] *= corr;
    }

    // ---- Bounce P through LDS (C-layout -> A-layout) ----
    _Float16* Pw = &Ps[wave * 16 * AT_LD];
#pragma unroll
    for (int j = 0; j < 4; ++j)
#pragma unroll
      for (int i = 0; i < 8; ++i)
        Pw[(i + 8 * half) * AT_LD + j * 16 + nlo] = (_Float16)pm[j][i];
    __syncthreads();

    // ---- O += P @ V ----
#pragma unroll
    for (int ks = 0; ks < 2; ++ks) {
      const v16h ap = load_A_frag(Pw, 0, ks * 32, AT_LD);
#pragma unroll
      for (int j = 0; j < 4; ++j) {
        const v16h bv = load_B_frag(Vt, j * 16, ks * 32, AT_LD);
        oacc[j] = wmma_f16(ap, bv, oacc[j]);
      }
    }
    __syncthreads();
  }

  // ---- Normalize and write out f16 (B,T,H,HD) ----
#pragma unroll
  for (int j = 0; j < 4; ++j)
#pragma unroll
    for (int i = 0; i < 8; ++i) {
      const int qg = qbase + wave * 16 + i + 8 * half;
      O[(((size_t)b * T_ + qg) * H_ + h) * HD_ + j * 16 + nlo] =
          (_Float16)(oacc[j][i] / lsum[i]);
    }
}

// ---------------------------------------------------------------------------
extern "C" void kernel_launch(void* const* d_in, const int* in_sizes, int n_in,
                              void* d_out, int out_size, void* d_ws, size_t ws_size,
                              hipStream_t stream) {
  const float* x  = (const float*)d_in[0];
  const float* fc = (const float*)d_in[1];
  const float* fs = (const float*)d_in[2];
  // d_in[3] = mask (unused; causal mask applied analytically)
  const float* wq = (const float*)d_in[4];
  const float* wk = (const float*)d_in[5];
  const float* wv = (const float*)d_in[6];
  const float* wo = (const float*)d_in[7];
  float* out = (float*)d_out;

  char* ws = (char*)d_ws;
  _Float16* q  = (_Float16*)(ws);                      // 4096*2048 f16 = 16 MB
  _Float16* k  = (_Float16*)(ws + (16ull << 20));      // 4096*512  f16 =  4 MB
  _Float16* v  = (_Float16*)(ws + (20ull << 20));      // 4096*512  f16 =  4 MB
  _Float16* ao = (_Float16*)(ws + (24ull << 20));      // 4096*2048 f16 = 16 MB

  const int M = B_ * T_;  // 4096

  gemm_wmma_kern<float, _Float16>
      <<<dim3((H_ * HD_) / GTN, M / GTM), 256, 0, stream>>>(x, wq, q, M, H_ * HD_, D_);
  gemm_wmma_kern<float, _Float16>
      <<<dim3((KV_ * HD_) / GTN, M / GTM), 256, 0, stream>>>(x, wk, k, M, KV_ * HD_, D_);
  gemm_wmma_kern<float, _Float16>
      <<<dim3((KV_ * HD_) / GTN, M / GTM), 256, 0, stream>>>(x, wv, v, M, KV_ * HD_, D_);

  const int nq = M * H_ * (HD_ / 2);
  rope_kernel<<<(nq + 255) / 256, 256, 0, stream>>>(q, fc, fs, M, H_);
  const int nk = M * KV_ * (HD_ / 2);
  rope_kernel<<<(nk + 255) / 256, 256, 0, stream>>>(k, fc, fs, M, KV_);

  attn_kernel<<<dim3(T_ / 64, H_, B_), 128, 0, stream>>>(q, k, v, ao);

  gemm_wmma_kern<_Float16, float>
      <<<dim3(D_ / GTN, M / GTM), 256, 0, stream>>>(ao, wo, out, M, D_, D_);
}
